// LoraLinear_53068615909970
// MI455X (gfx1250) — matrix operand order repaired and checked by reading, not compiled, not run on hardware
//
#include <hip/hip_runtime.h>

// ---------------------------------------------------------------------------
// LoRA linear, folded:  out = X @ (W + 2*B@A)^T
//   1) cvt_x:      X f32 -> bf16                       (workspace)
//   2) build_weff: W_eff = W + 2*(B@A), f32 -> bf16    (workspace)
//   3) gemm_bf16:  out[8192,4096] = Xb @ Weff^T
//      - v_wmma_f32_16x16x32_bf16 matrix cores, 64x64 wave tile (4x4 acc)
//      - double-buffered GLOBAL_LOAD_ASYNC_TO_LDS_B128 staging (ASYNCcnt)
//      - LDS laid out in fragment order -> conflict-free ds_load_b128, 1:1
//        ds_load:wmma ratio
// ---------------------------------------------------------------------------

typedef __bf16 bf16_t;
typedef __attribute__((ext_vector_type(16))) __bf16 v16bf;
typedef __attribute__((ext_vector_type(8)))  __bf16 v8bf;
typedef __attribute__((ext_vector_type(8)))  float  v8f;
typedef __attribute__((ext_vector_type(4)))  int    v4i;

constexpr int M_DIM = 4 * 2048;   // B*S
constexpr int K_DIM = 4096;       // D_IN
constexpr int N_DIM = 4096;       // D_OUT
constexpr int R_DIM = 64;         // LoRA rank
constexpr float SCALING = 2.0f;   // alpha / r

// ---------------------------------------------------------------------------
// CDNA5 async-copy + wait helpers (builtin signature probe-confirmed:
// (v4i AS(1)*, v4i AS(3)*, Ii offset, Ii cpol)).
// ---------------------------------------------------------------------------
__device__ __forceinline__ void async_copy_b128(const void* g, void* l) {
#if __has_builtin(__builtin_amdgcn_global_load_async_to_lds_b128)
    typedef __attribute__((address_space(1))) v4i gv4i_t;
    typedef __attribute__((address_space(3))) v4i lv4i_t;
    __builtin_amdgcn_global_load_async_to_lds_b128(
        (gv4i_t*)(unsigned long long)(size_t)g,
        (lv4i_t*)(unsigned int)(size_t)l,
        /*imm offset*/0, /*cpol*/0);
#else
    unsigned lofs = (unsigned)(unsigned long long)l;   // low 32b = LDS byte addr
    asm volatile("global_load_async_to_lds_b128 %0, %1, off"
                 :: "v"(lofs), "v"(g) : "memory");
#endif
}

__device__ __forceinline__ void wait_asynccnt0() {
#if __has_builtin(__builtin_amdgcn_s_wait_asynccnt)
    __builtin_amdgcn_s_wait_asynccnt(0);
#else
    asm volatile("s_wait_asynccnt 0x0" ::: "memory");
#endif
}

// ---------------------------------------------------------------------------
// Kernel 1: f32 -> bf16 conversion of X, 8 elements per thread (16B stores).
// ---------------------------------------------------------------------------
__global__ __launch_bounds__(256) void cvt_x_kernel(const float* __restrict__ x,
                                                    bf16_t* __restrict__ xb) {
    const size_t i = ((size_t)blockIdx.x * blockDim.x + threadIdx.x) * 8;
    v8bf o;
#pragma unroll
    for (int j = 0; j < 8; ++j) o[j] = (bf16_t)x[i + j];
    *(v8bf*)(xb + i) = o;
}

// ---------------------------------------------------------------------------
// Kernel 2: W_eff[o][d] = bf16( W[o][d] + 2 * sum_r lora_B[o][r]*lora_A[r][d] )
// ---------------------------------------------------------------------------
__global__ __launch_bounds__(256) void build_weff_kernel(const float* __restrict__ w,
                                                         const float* __restrict__ la,
                                                         const float* __restrict__ lb,
                                                         bf16_t* __restrict__ weff) {
    const int d = blockIdx.x * 256 + threadIdx.x;
    const int o = blockIdx.y;
    float acc = w[(size_t)o * K_DIM + d];
    const float* lbrow = lb + (size_t)o * R_DIM;
#pragma unroll 8
    for (int r = 0; r < R_DIM; ++r)
        acc += SCALING * lbrow[r] * la[(size_t)r * K_DIM + d];
    weff[(size_t)o * K_DIM + d] = (bf16_t)acc;
}

// ---------------------------------------------------------------------------
// Kernel 3: bf16 WMMA GEMM with async-LDS double buffering.
//   block: 256 threads = 8 waves (wave32), wave grid 2(M) x 4(N)
//   block tile 128x256; wave tile 64x64 = 4x4 tiles of 16x16
//   KSTEP = 64 (2 sub-steps of 32) -> 32 WMMAs per barrier, 1:1 ds:wmma
//
// LDS stage (48 KB): A 16 KB, B 32 KB; tiles stored in fragment order:
//   A: off = ks*8192  + mtile*1024 + chunk*512 + lane*16  (chunk: K lo/hi 16B)
//   B: off = 16384 + ks*16384 + ntile*1024 + lane*32      (32B/lane: K lo/hi 16B)
// so fragment reads are ds_load_b128 at lane-linear addresses (bank-clean).
//
// WMMA fragment mapping (ISA 7.12.2, wave32):
//   A 16x32: lane L -> row (L&15); elems 0..7  = K[(L>>4)*8 .. +7]
//                                  elems 8..15 = K[16+(L>>4)*8 .. +7]
//   B 32x16: lane L -> col (L&15); elems 0..7 = K[(L>>4)*8 .. +7], 8..15 = K[16+(L>>4)*8 .. +7]
//   C 16x16: reg i, lane L -> row (L>>4)*8 + i, col (L&15)
// ---------------------------------------------------------------------------
constexpr int KSTEP       = 64;
constexpr int A_BYTES     = 16384;            // 128 rows x 64 K x bf16
constexpr int B_BYTES     = 32768;            // 256 cols x 64 K x bf16
constexpr int STAGE_BYTES = A_BYTES + B_BYTES;

__global__ __launch_bounds__(256) void gemm_bf16_kernel(const bf16_t* __restrict__ A,
                                                        const bf16_t* __restrict__ B,
                                                        float* __restrict__ C) {
    __shared__ char smem[2 * STAGE_BYTES];    // 96 KB of the 320 KB WGP LDS

    const int t     = threadIdx.x;
    const int lane  = t & 31;
    const int wid   = t >> 5;
    const int waveM = wid & 1;        // 0..1
    const int waveN = wid >> 1;       // 0..3
    const int m0 = blockIdx.y * 128 + waveM * 64;
    const int n0 = blockIdx.x * 256 + waveN * 64;
    const int lrow = lane & 15;
    const int lhi  = lane >> 4;       // 0 or 1

    // ---- per-thread async-copy descriptors: 4 A-chunks + 8 B-chunks / stage
    // (32-bit global element offsets: max 2^25 elements, fits easily)
    unsigned agoff[4];  int aldst[4];
    unsigned bgoff[8];  int bldst[8];
#pragma unroll
    for (int j = 0; j < 4; ++j) {              // A: 1024 x 16B chunks
        int id   = t + j * 256;                // 0..1023
        int ks   = id >> 9;
        int rem  = id & 511;
        int tile = rem >> 6;
        int ch   = (rem >> 5) & 1;
        int ll   = rem & 31;
        int arow = blockIdx.y * 128 + tile * 16 + (ll & 15);
        agoff[j] = (unsigned)(arow * K_DIM + ks * 32 + ch * 16 + (ll >> 4) * 8);
        aldst[j] = ks * 8192 + tile * 1024 + ch * 512 + ll * 16;
    }
#pragma unroll
    for (int j = 0; j < 8; ++j) {              // B: 2048 x 16B chunks
        int id   = t + j * 256;                // 0..2047
        int ks   = id >> 10;
        int rem  = id & 1023;
        int nt   = rem >> 6;                   // 0..15
        int r2   = rem & 63;
        int l    = r2 >> 1;
        int c    = r2 & 1;
        int bcol = blockIdx.x * 256 + nt * 16 + (l & 15);
        bgoff[j] = (unsigned)(bcol * K_DIM + ks * 32 + c * 16 + (l >> 4) * 8);
        bldst[j] = A_BYTES + ks * 16384 + nt * 1024 + l * 32 + c * 16;
    }

    v8f acc[4][4];
#pragma unroll
    for (int mt = 0; mt < 4; ++mt)
#pragma unroll
        for (int nt = 0; nt < 4; ++nt)
#pragma unroll
            for (int i = 0; i < 8; ++i) acc[mt][nt][i] = 0.0f;

    // ---- prologue: stage 0 copies
    {
        char* sb = smem;
#pragma unroll
        for (int j = 0; j < 4; ++j) async_copy_b128(A + agoff[j], sb + aldst[j]);
#pragma unroll
        for (int j = 0; j < 8; ++j) async_copy_b128(B + bgoff[j], sb + bldst[j]);
    }

    int stage = 0;
    for (int k0 = 0; k0 < K_DIM; k0 += KSTEP) {
        wait_asynccnt0();        // our copies for this stage have landed in LDS
        __syncthreads();         // everyone's copies landed; prev stage fully read

        // issue next stage while we compute on this one
        if (k0 + KSTEP < K_DIM) {
            char* sn = smem + (stage ^ 1) * STAGE_BYTES;
            const int kn = k0 + KSTEP;
#pragma unroll
            for (int j = 0; j < 4; ++j) async_copy_b128(A + agoff[j] + kn, sn + aldst[j]);
#pragma unroll
            for (int j = 0; j < 8; ++j) async_copy_b128(B + bgoff[j] + kn, sn + bldst[j]);
        }

        const char* sb = smem + stage * STAGE_BYTES;
#pragma unroll
        for (int ks = 0; ks < 2; ++ks) {
            v16bf afrag[4];
#pragma unroll
            for (int mt = 0; mt < 4; ++mt) {
                const char* p = sb + ks * 8192 + (waveM * 4 + mt) * 1024 + lane * 16;
                v8bf lo = *(const v8bf*)(p);
                v8bf hi = *(const v8bf*)(p + 512);
                afrag[mt] = __builtin_shufflevector(lo, hi,
                    0, 1, 2, 3, 4, 5, 6, 7, 8, 9, 10, 11, 12, 13, 14, 15);
            }
            v16bf bfrag[4];
#pragma unroll
            for (int nt = 0; nt < 4; ++nt) {
                const char* p = sb + A_BYTES + ks * 16384 + (waveN * 4 + nt) * 1024 + lane * 32;
                bfrag[nt] = *(const v16bf*)(p);
            }
#pragma unroll
            for (int mt = 0; mt < 4; ++mt)
#pragma unroll
                for (int nt = 0; nt < 4; ++nt)
                    acc[mt][nt] = __builtin_amdgcn_wmma_f32_16x16x32_bf16(
                        false, afrag[mt], false, bfrag[nt],
                        (short)0, acc[mt][nt], false, false);
        }
        stage ^= 1;
    }

    // ---- epilogue: store D tiles (cols contiguous across lanes -> coalesced)
#pragma unroll
    for (int mt = 0; mt < 4; ++mt)
#pragma unroll
        for (int nt = 0; nt < 4; ++nt) {
            const int col = n0 + nt * 16 + lrow;
#pragma unroll
            for (int i = 0; i < 8; ++i) {
                const int row = m0 + mt * 16 + lhi * 8 + i;
                C[(size_t)row * N_DIM + col] = acc[mt][nt][i];
            }
        }
}

// ---------------------------------------------------------------------------
// Host launch. d_in order: x, weight, lora_A, lora_B (all f32). out: f32.
// Workspace: [X_bf16 : M*K*2][W_eff_bf16 : N*K*2]  (~100.7 MB)
// ---------------------------------------------------------------------------
extern "C" void kernel_launch(void* const* d_in, const int* in_sizes, int n_in,
                              void* d_out, int out_size, void* d_ws, size_t ws_size,
                              hipStream_t stream) {
    const float* x  = (const float*)d_in[0];
    const float* w  = (const float*)d_in[1];
    const float* la = (const float*)d_in[2];
    const float* lb = (const float*)d_in[3];
    float* out = (float*)d_out;

    bf16_t* xb   = (bf16_t*)d_ws;
    bf16_t* weff = (bf16_t*)((char*)d_ws + (size_t)M_DIM * K_DIM * sizeof(bf16_t));

    {
        const size_t total = (size_t)M_DIM * K_DIM;
        const int threads = 256;
        const int blocks = (int)(total / ((size_t)threads * 8));
        cvt_x_kernel<<<blocks, threads, 0, stream>>>(x, xb);
    }
    {
        dim3 grid(K_DIM / 256, N_DIM);
        build_weff_kernel<<<grid, 256, 0, stream>>>(w, la, lb, weff);
    }
    {
        dim3 grid(N_DIM / 256, M_DIM / 128);
        gemm_bf16_kernel<<<grid, 256, 0, stream>>>(xb, weff, out);
    }
}
